// VectorQuantizer_23433341567670
// MI455X (gfx1250) — matrix operand order repaired and checked by reading, not compiled, not run on hardware
//
#include <hip/hip_runtime.h>

typedef __attribute__((ext_vector_type(16))) _Float16 v16h;
typedef __attribute__((ext_vector_type(8)))  float    v8f;
typedef __attribute__((ext_vector_type(2)))  float    f32x2;
typedef __attribute__((ext_vector_type(4)))  float    f32x4;

#define DDIM   64
#define KCODES 1024
#define NROWS  131072
#define CBLK   64                  // 1024 / 16 column blocks
#define TILES  2                   // 16-row tiles per wave
#define WAVES  8
#define ROWS_PER_WG (16*TILES*WAVES)   // 256
#define FRAG_DWORDS (CBLK*2*32*8)      // 32768 dwords = 128 KB of B fragments
#define DIST_BIAS 1024.0f              // keeps e_sq - 2*cross strictly positive

// ---------------------------------------------------------------------------
// Prep 1: swizzle embeddings (f32, row-major KxD) into f16 WMMA B-fragments.
// frag[((cb*2+kb)*32 + L)*8 + v] = pack_f16(E[cb*16+(L&15)][kb*32+k0],
//                                           E[cb*16+(L&15)][kb*32+k0+1])
// lane L holds B-column n=L&15, dword v holds k0=((v>>2)<<4)+(L>>4)*8+((v&3)<<1), k0+1.
// ---------------------------------------------------------------------------
__global__ __launch_bounds__(256) void vq_prep_frag(const float* __restrict__ E,
                                                    unsigned* __restrict__ frag) {
  int tid = blockIdx.x * 256 + threadIdx.x;    // 0 .. 65535
  int v   = tid & 7;
  int L   = (tid >> 3) & 31;
  int kb  = (tid >> 8) & 1;
  int cb  = tid >> 9;
  int d   = kb * 32 + ((v >> 2) << 4) + (L >> 4) * 8 + ((v & 3) << 1);
  int n0  = cb * 16 + (L & 15);
  _Float16 lo = (_Float16)E[n0 * DDIM + d];
  _Float16 hi = (_Float16)E[n0 * DDIM + d + 1];
  unsigned w = ((unsigned)*(const unsigned short*)&hi << 16) |
                (unsigned)*(const unsigned short*)&lo;
  frag[tid] = w;    // tid == ((cb*2+kb)*32 + L)*8 + v by construction
}

// ---------------------------------------------------------------------------
// Prep 2: esq[k] = sum_d E[k][d]^2 + DIST_BIAS  (bias folded in for free so
// the per-candidate distance key is always a positive float -> uint-monotone).
// ---------------------------------------------------------------------------
__global__ __launch_bounds__(256) void vq_prep_esq(const float* __restrict__ E,
                                                   float* __restrict__ esq) {
  int k = blockIdx.x * 256 + threadIdx.x;      // 0 .. 1023
  const float* row = E + (long)k * DDIM;
  float s = 0.f;
#pragma unroll 8
  for (int d = 0; d < DDIM; ++d) { float e = row[d]; s += e * e; }
  esq[k] = s + DIST_BIAS;
}

// ---------------------------------------------------------------------------
// Main kernel: 8 waves * 2 tiles = 256 rows per workgroup.
// ---------------------------------------------------------------------------
__global__ __launch_bounds__(256) void vq_main(const float* __restrict__ x,
                                               const float* __restrict__ E,
                                               const unsigned* __restrict__ frag,
                                               const float* __restrict__ esq,
                                               float* __restrict__ out) {
  extern __shared__ char smem[];
  unsigned* sFrag = (unsigned*)smem;                       // 128 KB
  float*    sEsq  = (float*)(smem + FRAG_DWORDS * 4);      //   4 KB

  const int tid = threadIdx.x;

  // --- Stage B fragments + e_sq into LDS via async global->LDS DMA ---------
  {
    unsigned ldsFragBase = (unsigned)(uintptr_t)(void*)sFrag;
    unsigned ldsEsqBase  = (unsigned)(uintptr_t)(void*)sEsq;
#pragma unroll
    for (int i = 0; i < (FRAG_DWORDS / 4) / 256; ++i) {     // 32 chunks/thread
      int idx = i * 256 + tid;                              // b128 chunk id
      unsigned lds_off = ldsFragBase + (unsigned)idx * 16u;
      unsigned long long gaddr =
          (unsigned long long)(uintptr_t)((const char*)frag + (size_t)idx * 16);
      asm volatile("global_load_async_to_lds_b128 %0, %1, off"
                   :: "v"(lds_off), "v"(gaddr) : "memory");
    }
    {                                                       // 4 KB of e_sq
      unsigned lds_off = ldsEsqBase + (unsigned)tid * 16u;
      unsigned long long gaddr =
          (unsigned long long)(uintptr_t)((const char*)esq + (size_t)tid * 16);
      asm volatile("global_load_async_to_lds_b128 %0, %1, off"
                   :: "v"(lds_off), "v"(gaddr) : "memory");
    }
    asm volatile("s_wait_asynccnt 0" ::: "memory");
  }
  __syncthreads();

  const int lane = tid & 31;
  const int wave = tid >> 5;
  const int h    = lane >> 4;      // half-wave selector (A layout / C row+8)
  const int rl   = lane & 15;      // row-in-tile for A, code column for C
  const long rowBase = (long)blockIdx.x * ROWS_PER_WG + (long)wave * (16 * TILES);

  // --- A fragments: 16x32 f16 layout (ISA 7.12.2), two k-blocks per tile ---
  v16h A[TILES][2];
#pragma unroll
  for (int t = 0; t < TILES; ++t) {
    const long row = rowBase + t * 16 + rl;
    const f32x2* xr = (const f32x2*)(x + row * DDIM);
#pragma unroll
    for (int kb = 0; kb < 2; ++kb) {
#pragma unroll
      for (int v = 0; v < 8; ++v) {
        int k0 = kb * 32 + ((v >> 2) << 4) + h * 8 + ((v & 3) << 1);
        f32x2 f = __builtin_nontemporal_load(xr + (k0 >> 1));
        A[t][kb][2 * v]     = (_Float16)f.x;
        A[t][kb][2 * v + 1] = (_Float16)f.y;
      }
    }
  }

  // Running best as packed keys: bits[31:6] = quantized positive dist,
  // bits[5:0] = column-block id. Single v_and_or_b32 + v_min_u32 per cand.
  unsigned bkey[TILES][8];
#pragma unroll
  for (int t = 0; t < TILES; ++t)
#pragma unroll
    for (int r = 0; r < 8; ++r) bkey[t][r] = 0xFFFFFFFFu;

  // --- Sweep all 64 column blocks of 16 codes (unroll x2: hides the
  //     WMMA->VALU hazard slots and halves SALU loop overhead) ---
#pragma unroll 2
  for (int cb = 0; cb < CBLK; ++cb) {
    const float en = sEsq[cb * 16 + rl];       // e_sq + bias
    const v16h B0 = *(const v16h*)&sFrag[((cb * 2 + 0) * 32 + lane) * 8];
    const v16h B1 = *(const v16h*)&sFrag[((cb * 2 + 1) * 32 + lane) * 8];
#pragma unroll
    for (int t = 0; t < TILES; ++t) {
      v8f c = {};
      c = __builtin_amdgcn_wmma_f32_16x16x32_f16(false, A[t][0], false, B0,
                                                 (short)0, c, false, false);
      c = __builtin_amdgcn_wmma_f32_16x16x32_f16(false, A[t][1], false, B1,
                                                 (short)0, c, false, false);
#pragma unroll
      for (int r = 0; r < 8; ++r) {
        float dc = __builtin_fmaf(c[r], -2.0f, en);   // bias + esq - 2*cross
        unsigned key = (__float_as_uint(dc) & 0xFFFFFFC0u) | (unsigned)cb;
        bkey[t][r] = key < bkey[t][r] ? key : bkey[t][r];
      }
    }
  }

  // --- Min+argmin across the 16 code lanes, then gather E[idx] -> out ---
#pragma unroll
  for (int t = 0; t < TILES; ++t) {
#pragma unroll
    for (int r = 0; r < 8; ++r) {
      unsigned kv = bkey[t][r];
      int      ix = (int)((kv & 63u) << 4) | rl;      // cb*16 + column
#pragma unroll
      for (int m = 8; m >= 1; m >>= 1) {
        unsigned ok = (unsigned)__shfl_xor((int)kv, m, 16);
        int      oi = __shfl_xor(ix, m, 16);
        if (ok < kv || (ok == kv && oi < ix)) { kv = ok; ix = oi; }
      }
      const long row = rowBase + t * 16 + r + h * 8;
      const f32x4* src = (const f32x4*)(E + (long)ix * DDIM);   // L2-hot gather
      f32x4*       dst = (f32x4*)(out + row * DDIM);
      __builtin_nontemporal_store(src[rl], dst + rl);
    }
  }
}

// ---------------------------------------------------------------------------
extern "C" void kernel_launch(void* const* d_in, const int* in_sizes, int n_in,
                              void* d_out, int out_size, void* d_ws, size_t ws_size,
                              hipStream_t stream) {
  const float* x = (const float*)d_in[0];       // (131072, 64) f32
  const float* E = (const float*)d_in[1];       // (1024, 64) f32
  float* out = (float*)d_out;                   // (131072, 64) f32

  unsigned* frag = (unsigned*)d_ws;                          // 128 KB
  float*    esq  = (float*)((char*)d_ws + FRAG_DWORDS * 4);  //   4 KB

  const size_t shmem = FRAG_DWORDS * 4 + KCODES * 4;         // 132 KB
  (void)hipFuncSetAttribute((const void*)vq_main,
                            hipFuncAttributeMaxDynamicSharedMemorySize,
                            (int)shmem);

  vq_prep_frag<<<(KCODES * DDIM) / 256, 256, 0, stream>>>(E, frag);
  vq_prep_esq<<<KCODES / 256, 256, 0, stream>>>(E, esq);
  vq_main<<<NROWS / ROWS_PER_WG, 256, shmem, stream>>>(x, E, frag, esq, out);
}
